// GraphRLNetwork_74586402062932
// MI455X (gfx1250) — matrix operand order repaired and checked by reading, not compiled, not run on hardware
//
#include <hip/hip_runtime.h>
#include <stdint.h>

typedef __attribute__((ext_vector_type(16))) _Float16 v16h;
typedef __attribute__((ext_vector_type(8)))  _Float16 v8h;
typedef __attribute__((ext_vector_type(8)))  float    v8f;

#define N_NODES 10000
#define N_EDGES 320000
#define HEADS   4
#define HID     128
#define HC      (HEADS * HID)   // 512
#define NEG_SLOPE 0.2f

// ---------------------------------------------------------------------------
// float atomic max via signed-max / unsigned-min bit trick (init to -inf).
// Lowers to GLOBAL_ATOMIC_MAX_I32 / GLOBAL_ATOMIC_MIN_U32 (native on gfx1250).
// ---------------------------------------------------------------------------
__device__ __forceinline__ void atomicMaxFloat(float* addr, float val) {
    if (val >= 0.0f) {
        atomicMax((int*)addr, __float_as_int(val));
    } else {
        atomicMin((unsigned int*)addr, __float_as_uint(val));
    }
}

// ---------------------------------------------------------------------------
// Pack activations [Nr x K] f32 -> f16 row-major (8 elements / thread).
// ---------------------------------------------------------------------------
__global__ void pack_a_kernel(const float* __restrict__ A,
                              _Float16* __restrict__ Ah, int total8) {
    int t = blockIdx.x * blockDim.x + threadIdx.x;
    if (t >= total8) return;
    const float4 f0 = ((const float4*)A)[(size_t)t * 2];
    const float4 f1 = ((const float4*)A)[(size_t)t * 2 + 1];
    v8h o;
    o[0] = (_Float16)f0.x; o[1] = (_Float16)f0.y;
    o[2] = (_Float16)f0.z; o[3] = (_Float16)f0.w;
    o[4] = (_Float16)f1.x; o[5] = (_Float16)f1.y;
    o[6] = (_Float16)f1.z; o[7] = (_Float16)f1.w;
    ((v8h*)Ah)[t] = o;
}

// ---------------------------------------------------------------------------
// Pack B [K x M] f32 (row-major) into WMMA 16x16x32 f16 B-fragment order.
// Fragment layout (wave32): lane L holds N-col = tile_n*16 + (L&15),
// halves i=0..7 -> K = kt*32 + (L<16 ? 0 : 8) + i ; halves 8..15 -> +16 - 8 + i.
// Packed index: ((kt*tilesN + tn)*32 + lane)*16 + i   (tn-contiguous per kt)
// ---------------------------------------------------------------------------
__global__ void pack_b_kernel(const float* __restrict__ B, v16h* __restrict__ Bp,
                              int K, int M) {
    const int tilesN = M >> 4;
    const int ksteps = K >> 5;
    int t = blockIdx.x * blockDim.x + threadIdx.x;
    if (t >= ksteps * tilesN * 32) return;
    const int lane = t & 31;
    const int tn   = (t >> 5) % tilesN;
    const int kt   = (t >> 5) / tilesN;
    const int col   = tn * 16 + (lane & 15);
    const int kbase = kt * 32 + ((lane < 16) ? 0 : 8);
    v16h v;
#pragma unroll
    for (int i = 0; i < 16; ++i)
        v[i] = (_Float16)B[(size_t)(kbase + i + ((i >= 8) ? 8 : 0)) * M + col];
    Bp[t] = v;
}

// ---------------------------------------------------------------------------
// C[Nr x M] = act(A @ B + bias) via V_WMMA_F32_16X16X32_F16.
// Block = 8 waves sharing one NT-wide column-tile group. B fragments are
// double-buffered through LDS: stage kt+1 while WMMAs consume kt -> one
// barrier per K-step, global->LDS stage overlapped with the matrix pipe.
// Out-of-range row tiles are clamped (duplicate identical work) so every wave
// participates in barriers and WMMA executes with EXEC all-1s.
// A fragment (16-bit A 16x32 ISA layout): lane holds row tm*16+(lane&15);
// halves 0..7 -> K = kt*32 + (lane<16?0:8) + i ; halves 8..15 -> +16.
// ---------------------------------------------------------------------------
template <int NT>
__global__ void wmma_gemm_kernel(const _Float16* __restrict__ Ah,
                                 const v16h*  __restrict__ Bp,
                                 const float* __restrict__ bias,
                                 float* __restrict__ C,
                                 int Nrows, int K, int M, int relu) {
    __shared__ v16h Bsh[2][NT * 32];

    const int lane    = threadIdx.x & 31;
    const int wave    = threadIdx.x >> 5;
    const int tilesN  = M >> 4;
    const int tilesM  = Nrows >> 4;
    const int ngroups = tilesN / NT;
    const int tng     = blockIdx.x % ngroups;
    int tm            = (blockIdx.x / ngroups) * 8 + wave;
    if (tm >= tilesM) tm = tilesM - 1;           // clamp: benign duplicate tile

    const int row  = tm * 16 + (lane & 15);
    const int koff = (lane < 16) ? 0 : 8;
    const _Float16* __restrict__ Arow = Ah + (size_t)row * K;
    const v16h* __restrict__ Bgrp = Bp + (size_t)tng * NT * 32;

    const int ksteps = K >> 5;

    // prologue: stage kt = 0
    if (threadIdx.x < NT * 32)
        Bsh[0][threadIdx.x] = Bgrp[threadIdx.x];
    __syncthreads();

    v8f acc[NT] = {};
    for (int kt = 0; kt < ksteps; ++kt) {
        const int cur = kt & 1;
        // stage next B panel into the other buffer (overlaps with WMMAs below)
        if (kt + 1 < ksteps && threadIdx.x < NT * 32)
            Bsh[cur ^ 1][threadIdx.x] =
                Bgrp[(size_t)(kt + 1) * tilesN * 32 + threadIdx.x];

        // A fragment: two aligned 16B chunks, already f16
        const int kb = kt * 32 + koff;
        const v8h lo = *(const v8h*)(Arow + kb);
        const v8h hi = *(const v8h*)(Arow + kb + 16);
        const v16h a = __builtin_shufflevector(lo, hi,
            0, 1, 2, 3, 4, 5, 6, 7, 8, 9, 10, 11, 12, 13, 14, 15);
        if (kt + 1 < ksteps)
            __builtin_prefetch(Arow + kb + 32, 0, 3);   // global_prefetch_b8

#pragma unroll
        for (int j = 0; j < NT; ++j) {
            acc[j] = __builtin_amdgcn_wmma_f32_16x16x32_f16(
                /*neg_a=*/false, a, /*neg_b=*/false, Bsh[cur][j * 32 + lane],
                /*c_mod=*/(short)0, acc[j],
                /*reuse_a=*/false, /*reuse_b=*/false);
        }
        __syncthreads();   // next buffer staged & current buffer consumed
    }

    // Epilogue: C/D layout -> lane holds col = (lane&15), rows v+(lane<16?0:8)
    const int row0 = tm * 16 + ((lane < 16) ? 0 : 8);
#pragma unroll
    for (int j = 0; j < NT; ++j) {
        const int col = (tng * NT + j) * 16 + (lane & 15);
        const float bv = bias ? bias[col] : 0.0f;
        float* __restrict__ Cp = C + (size_t)row0 * M + col;
#pragma unroll
        for (int v = 0; v < 8; ++v) {
            float r = acc[j][v] + bv;
            if (relu) r = fmaxf(r, 0.0f);
            Cp[(size_t)v * M] = r;
        }
    }
}

// ---------------------------------------------------------------------------
// Per-node attention scores: e_s[n,h] = <hW[n,h,:], a_src[h,:]>, e_d likewise.
// ---------------------------------------------------------------------------
__global__ void node_scores_kernel(const float* __restrict__ hW,
                                   const float* __restrict__ a_src,
                                   const float* __restrict__ a_dst,
                                   float* __restrict__ es, float* __restrict__ ed) {
    int t = blockIdx.x * blockDim.x + threadIdx.x;
    if (t >= N_NODES * HEADS) return;
    const int n = t >> 2, h = t & 3;
    const float* __restrict__ hp = hW + (size_t)n * HC + h * HID;
    const float* __restrict__ as = a_src + h * HID;
    const float* __restrict__ ad = a_dst + h * HID;
    float s1 = 0.0f, s2 = 0.0f;
    for (int c = 0; c < HID; c += 4) {
        float4 hv = *(const float4*)(hp + c);
        float4 av = *(const float4*)(as + c);
        float4 dv = *(const float4*)(ad + c);
        s1 += hv.x * av.x + hv.y * av.y + hv.z * av.z + hv.w * av.w;
        s2 += hv.x * dv.x + hv.y * dv.y + hv.z * dv.z + hv.w * dv.w;
    }
    es[t] = s1;
    ed[t] = s2;
}

// m = -inf, denom = 0, agg = 0
__global__ void init_layer_kernel(float* __restrict__ m, float* __restrict__ den,
                                  float* __restrict__ agg) {
    int t = blockIdx.x * blockDim.x + threadIdx.x;
    if (t < N_NODES * HEADS) { m[t] = -__builtin_inff(); den[t] = 0.0f; }
    if (t < N_NODES * HC)    agg[t] = 0.0f;
}

__global__ void edge_max_kernel(const long long* __restrict__ src,
                                const long long* __restrict__ dst,
                                const float* __restrict__ es,
                                const float* __restrict__ ed,
                                float* __restrict__ m) {
    int t = blockIdx.x * blockDim.x + threadIdx.x;
    if (t >= N_EDGES * HEADS) return;
    const int e = t >> 2, h = t & 3;
    const int s = (int)src[e], d = (int)dst[e];
    float sc = es[s * HEADS + h] + ed[d * HEADS + h];
    sc = (sc > 0.0f) ? sc : NEG_SLOPE * sc;
    atomicMaxFloat(&m[d * HEADS + h], sc);
}

__global__ void edge_denom_kernel(const long long* __restrict__ src,
                                  const long long* __restrict__ dst,
                                  const float* __restrict__ es,
                                  const float* __restrict__ ed,
                                  const float* __restrict__ m,
                                  float* __restrict__ den) {
    int t = blockIdx.x * blockDim.x + threadIdx.x;
    if (t >= N_EDGES * HEADS) return;
    const int e = t >> 2, h = t & 3;
    const int s = (int)src[e], d = (int)dst[e];
    float sc = es[s * HEADS + h] + ed[d * HEADS + h];
    sc = (sc > 0.0f) ? sc : NEG_SLOPE * sc;
    float mm = m[d * HEADS + h];
    if (mm < -3.0e38f) mm = 0.0f;               // isfinite guard (ref semantics)
    atomicAdd(&den[d * HEADS + h], __expf(sc - mm));
}

// One thread per (edge, 4-channel chunk): recompute alpha, scatter-add message.
__global__ void edge_agg_kernel(const long long* __restrict__ src,
                                const long long* __restrict__ dst,
                                const float* __restrict__ es,
                                const float* __restrict__ ed,
                                const float* __restrict__ m,
                                const float* __restrict__ den,
                                const float* __restrict__ hW,
                                float* __restrict__ agg) {
    int t = blockIdx.x * blockDim.x + threadIdx.x;
    if (t >= N_EDGES * (HC / 4)) return;
    const int e     = t >> 7;                   // 128 chunks of 4 per edge
    const int chunk = t & 127;
    const int hc0   = chunk * 4;
    const int h     = hc0 >> 7;                 // hc0 / HID
    const int s = (int)src[e];
    const int d = (int)dst[e];
    float sc = es[s * HEADS + h] + ed[d * HEADS + h];
    sc = (sc > 0.0f) ? sc : NEG_SLOPE * sc;
    float mm = m[d * HEADS + h];
    if (mm < -3.0e38f) mm = 0.0f;
    const float alpha = __expf(sc - mm) / (den[d * HEADS + h] + 1e-16f);
    const float4 hv = *(const float4*)(hW + (size_t)s * HC + hc0);
    float* __restrict__ op = agg + (size_t)d * HC + hc0;
    atomicAdd(op + 0, alpha * hv.x);
    atomicAdd(op + 1, alpha * hv.y);
    atomicAdd(op + 2, alpha * hv.z);
    atomicAdd(op + 3, alpha * hv.w);
}

__global__ void bias_relu_kernel(float* __restrict__ agg,
                                 const float* __restrict__ b, int total, int M) {
    int t = blockIdx.x * blockDim.x + threadIdx.x;
    if (t >= total) return;
    agg[t] = fmaxf(agg[t] + b[t % M], 0.0f);
}

// ---------------------------------------------------------------------------
extern "C" void kernel_launch(void* const* d_in, const int* in_sizes, int n_in,
                              void* d_out, int out_size, void* d_ws, size_t ws_size,
                              hipStream_t stream) {
    (void)in_sizes; (void)n_in; (void)out_size; (void)ws_size;
    const int N = N_NODES, E = N_EDGES;

    const float*      x     = (const float*)d_in[0];
    const long long*  ei    = (const long long*)d_in[1];   // int64 edge_index [2,E]
    const long long*  src   = ei;
    const long long*  dst   = ei + E;
    const float*      W_emb = (const float*)d_in[2];
    const float*      b_emb = (const float*)d_in[3];
    const float* W[3]    = {(const float*)d_in[4],  (const float*)d_in[8],  (const float*)d_in[12]};
    const float* asrc[3] = {(const float*)d_in[5],  (const float*)d_in[9],  (const float*)d_in[13]};
    const float* adst[3] = {(const float*)d_in[6],  (const float*)d_in[10], (const float*)d_in[14]};
    const float* bl[3]   = {(const float*)d_in[7],  (const float*)d_in[11], (const float*)d_in[15]};
    const float* Wo1 = (const float*)d_in[16];
    const float* bo1 = (const float*)d_in[17];
    const float* Wo2 = (const float*)d_in[18];
    const float* bo2 = (const float*)d_in[19];

    // workspace carve-out (256B aligned)
    char* ws = (char*)d_ws;
    auto carve = [&](size_t bytes) {
        char* p = ws;
        ws += (bytes + 255) & ~(size_t)255;
        return p;
    };
    float*    buf0  = (float*)carve((size_t)N * HC * 4);
    float*    buf1  = (float*)carve((size_t)N * HC * 4);
    float*    buf2  = (float*)carve((size_t)N * HC * 4);
    _Float16* Ah    = (_Float16*)carve((size_t)N * HC * 2);   // f16 activations
    v16h*     packB = (v16h*)  carve((size_t)HC * HC * 2);    // largest W 512x512
    float*    es    = (float*)carve((size_t)N * HEADS * 4);
    float*    edv   = (float*)carve((size_t)N * HEADS * 4);
    float*    mbuf  = (float*)carve((size_t)N * HEADS * 4);
    float*    den   = (float*)carve((size_t)N * HEADS * 4);

    auto gemm = [&](const float* A, const float* Bsrc, const float* bias, float* C,
                    int Nr, int K, int M, int relu) {
        const int total8 = (Nr * K) / 8;
        pack_a_kernel<<<(total8 + 255) / 256, 256, 0, stream>>>(A, Ah, total8);
        const int tilesN  = M / 16;
        const int pthread = (K / 32) * tilesN * 32;
        pack_b_kernel<<<(pthread + 255) / 256, 256, 0, stream>>>(Bsrc, packB, K, M);
        const int tilesM  = Nr / 16;
        if (tilesN % 8 == 0) {
            const int gridx = (tilesN / 8) * ((tilesM + 7) / 8);
            wmma_gemm_kernel<8><<<gridx, 256, 0, stream>>>(Ah, packB, bias, C,
                                                           Nr, K, M, relu);
        } else {
            const int gridx = (tilesN / 4) * ((tilesM + 7) / 8);
            wmma_gemm_kernel<4><<<gridx, 256, 0, stream>>>(Ah, packB, bias, C,
                                                           Nr, K, M, relu);
        }
    };

    // h = x @ W_emb + b_emb                          [N,128] -> buf0
    gemm(x, W_emb, b_emb, buf0, N, 64, HID, /*relu=*/0);

    float* h   = buf0;
    float* hw  = buf1;
    float* agg = buf2;
    const int inK[3] = {HID, HC, HC};

    for (int l = 0; l < 3; ++l) {
        // hW = h @ W[l]                              [N,512]
        gemm(h, W[l], nullptr, hw, N, inK[l], HC, /*relu=*/0);
        // per-node src/dst scores
        node_scores_kernel<<<(N * HEADS + 255) / 256, 256, 0, stream>>>(
            hw, asrc[l], adst[l], es, edv);
        // init m/denom/agg
        init_layer_kernel<<<(N * HC + 255) / 256, 256, 0, stream>>>(mbuf, den, agg);
        // segment softmax + aggregation
        edge_max_kernel<<<(E * HEADS + 255) / 256, 256, 0, stream>>>(src, dst, es, edv, mbuf);
        edge_denom_kernel<<<(E * HEADS + 255) / 256, 256, 0, stream>>>(src, dst, es, edv, mbuf, den);
        edge_agg_kernel<<<(E * (HC / 4) + 255) / 256, 256, 0, stream>>>(
            src, dst, es, edv, mbuf, den, hw, agg);
        // h_next = relu(agg + b[l])
        bias_relu_kernel<<<(N * HC + 255) / 256, 256, 0, stream>>>(agg, bl[l], N * HC, HC);
        // rotate buffers: new h = agg; recycle old h as hW; old hW as agg
        float* nh = agg; float* nhw = h; float* nagg = hw;
        h = nh; hw = nhw; agg = nagg;
    }

    // MLP head: hidden = relu(h @ Wo1 + bo1)         [N,128] -> hw (reused)
    gemm(h, Wo1, bo1, hw, N, HC, HID, /*relu=*/1);
    // out = hidden @ Wo2 + bo2                       [N,64]  -> d_out
    gemm(hw, Wo2, bo2, (float*)d_out, N, HID, 64, /*relu=*/0);
}
